// SageStreamModel_55688545960248
// MI455X (gfx1250) — matrix operand order repaired and checked by reading, not compiled, not run on hardware
//
#include <hip/hip_runtime.h>

typedef unsigned short u16;
typedef __attribute__((ext_vector_type(16))) __bf16 bf16x16;
typedef __attribute__((ext_vector_type(8)))  __bf16 bf16x8;
typedef __attribute__((ext_vector_type(8)))  float  f32x8;
typedef int v4i __attribute__((vector_size(16)));

#if defined(__has_builtin)
#if __has_builtin(__builtin_amdgcn_global_load_async_to_lds_b128) && \
    __has_builtin(__builtin_amdgcn_s_wait_asynccnt)
#define USE_ASYNC 1
#endif
#endif
#ifndef USE_ASYNC
#define USE_ASYNC 0
#endif

// ---------------- bf16 helpers ----------------
__device__ __forceinline__ float bf2f(u16 u) {
    union { unsigned int i; float f; } v; v.i = ((unsigned int)u) << 16; return v.f;
}
__device__ __forceinline__ u16 f2bf(float f) {
    union { float f; unsigned int i; } v; v.f = f;
    unsigned int x = v.i;
    unsigned int r = (x + 0x7FFFu + ((x >> 16) & 1u)) >> 16;  // RNE
    return (u16)r;
}

// per-lane 16B global -> LDS async copy (CDNA5 GLOBAL_LOAD_ASYNC_TO_LDS_B128)
__device__ __forceinline__ void async_cp16(const u16* g, u16* l) {
#if USE_ASYNC
    __builtin_amdgcn_global_load_async_to_lds_b128((v4i*)g, (v4i*)l, 0, 0);
#else
    (void)g; (void)l;
#endif
}
__device__ __forceinline__ void wait_async0() {
#if USE_ASYNC
    __builtin_amdgcn_s_wait_asynccnt(0);
#endif
}

// A-fragment (16-bit A 16x32): lane holds row M=lane&15, 16 consecutive K at kb=(lane>>4)*16
__device__ __forceinline__ bf16x16 frag_a(const u16* s, int base) {
    bf16x8 lo = *reinterpret_cast<const bf16x8*>(s + base);
    bf16x8 hi = *reinterpret_cast<const bf16x8*>(s + base + 8);
    return __builtin_shufflevector(lo, hi, 0,1,2,3,4,5,6,7,8,9,10,11,12,13,14,15);
}
// B-fragment from transposed LDS tile Bt[n][k]: lane col n=lane&15, 16 consecutive K at kb=(lane>>4)*16
__device__ __forceinline__ bf16x16 frag_b(const u16* s, int base) {
    bf16x8 lo = *reinterpret_cast<const bf16x8*>(s + base);
    bf16x8 hi = *reinterpret_cast<const bf16x8*>(s + base + 8);
    return __builtin_shufflevector(lo, hi, 0,1,2,3,4,5,6,7,8,9,10,11,12,13,14,15);
}
__device__ __forceinline__ f32x8 wmma_bf16(bf16x16 a, bf16x16 b, f32x8 c) {
    return __builtin_amdgcn_wmma_f32_16x16x32_bf16(false, a, false, b, (short)0, c, false, false);
}

#define LDT 40   // LDS row stride in halfs (80B, bank-conflict padding)

// ---------------- transpose + f32->bf16 conversion: src[n][R][C] -> dst[n][C][R] ----------------
__global__ __launch_bounds__(256) void convT_kernel(const float* __restrict__ src,
                                                    u16* __restrict__ dst,
                                                    int R, int C, long total) {
    long i = (long)blockIdx.x * 256 + threadIdx.x;
    if (i >= total) return;
    long per = (long)R * C;
    long n = i / per, rc = i % per;
    long r = rc / C, c = rc % C;
    dst[n * per + c * R + r] = f2bf(src[i]);
}

// ---------------- LayerNorm over D=512, one block (128 thr) per row ----------------
__global__ __launch_bounds__(128) void ln_kernel(const float* __restrict__ x,
                                                 const float* __restrict__ g,
                                                 const float* __restrict__ bb,
                                                 float* __restrict__ h) {
    __shared__ float red[128];
    const int n = blockIdx.x, t = threadIdx.x;
    const float* row = x + (size_t)n * 512;
    float v[4];
#pragma unroll
    for (int i = 0; i < 4; i++) v[i] = row[t + i * 128];
    float s = v[0] + v[1] + v[2] + v[3];
    red[t] = s; __syncthreads();
#pragma unroll
    for (int o = 64; o > 0; o >>= 1) { if (t < o) red[t] += red[t + o]; __syncthreads(); }
    float mean = red[0] * (1.f / 512.f);
    __syncthreads();
    float s2 = 0.f;
#pragma unroll
    for (int i = 0; i < 4; i++) { float d = v[i] - mean; s2 += d * d; }
    red[t] = s2; __syncthreads();
#pragma unroll
    for (int o = 64; o > 0; o >>= 1) { if (t < o) red[t] += red[t + o]; __syncthreads(); }
    float inv = rsqrtf(red[0] * (1.f / 512.f) + 1e-5f);
    float* hr = h + (size_t)n * 512;
#pragma unroll
    for (int i = 0; i < 4; i++) { int d = t + i * 128; hr[d] = (v[i] - mean) * inv * g[d] + bb[d]; }
}

// ---------------- per-(b,d) mean/var over T=1024 ----------------
__global__ __launch_bounds__(256) void colstats_kernel(const float* __restrict__ h,
                                                       float* __restrict__ mu,
                                                       float* __restrict__ rsig) {
    int b = blockIdx.x, dblk = blockIdx.y * 64;
    int tx = threadIdx.x & 63, ty = threadIdx.x >> 6;
    float s = 0.f, s2 = 0.f;
    const float* base = h + ((size_t)b * 1024) * 512 + dblk + tx;
    for (int t = ty; t < 1024; t += 4) { float v = base[(size_t)t * 512]; s += v; s2 += v * v; }
    __shared__ float rs[4][64], rq[4][64];
    rs[ty][tx] = s; rq[ty][tx] = s2; __syncthreads();
    if (ty == 0) {
        s  = rs[0][tx] + rs[1][tx] + rs[2][tx] + rs[3][tx];
        s2 = rq[0][tx] + rq[1][tx] + rq[2][tx] + rq[3][tx];
        float m = s * (1.f / 1024.f);
        float var = s2 * (1.f / 1024.f) - m * m;
        mu[b * 512 + dblk + tx] = m;
        rsig[b * 512 + dblk + tx] = rsqrtf(var + 1e-8f);
    }
}

// ---------------- subject-conditioning MLP: gamma/beta per (b,d) ----------------
__global__ __launch_bounds__(256) void subj_kernel(const int* __restrict__ sid,
                                                   const float* __restrict__ emb,
                                                   const float* __restrict__ h1w, const float* __restrict__ h1b,
                                                   const float* __restrict__ h2w, const float* __restrict__ h2b,
                                                   const float* __restrict__ s1w, const float* __restrict__ s1b,
                                                   const float* __restrict__ s2w, const float* __restrict__ s2b,
                                                   float* __restrict__ gamma, float* __restrict__ beta) {
    __shared__ float e_s[8 * 64], t1[8 * 128], t2[8 * 128], t3[8 * 64];
    int tid = threadIdx.x;
    for (int i = tid; i < 512; i += 256) { int b = i >> 6, j = i & 63; e_s[i] = emb[(size_t)sid[b] * 64 + j]; }
    __syncthreads();
    for (int i = tid; i < 1024; i += 256) {
        int b = i >> 7, j = i & 127; float a = h1b[j];
        for (int k = 0; k < 64; k++) a += e_s[b * 64 + k] * h1w[k * 128 + j];
        t1[i] = a > 0.f ? a : 0.f;
    }
    __syncthreads();
    for (int i = tid; i < 1024; i += 256) {
        int b = i >> 7, j = i & 127; float a = h2b[j];
        for (int k = 0; k < 128; k++) a += t1[b * 128 + k] * h2w[k * 128 + j];
        t2[i] = a;
    }
    __syncthreads();
    for (int i = tid; i < 512; i += 256) {
        int b = i >> 6, j = i & 63; float a = s1b[j];
        for (int k = 0; k < 128; k++) a += t2[b * 128 + k] * s1w[k * 64 + j];
        t3[i] = a > 0.f ? a : 0.f;
    }
    __syncthreads();
    for (int i = tid; i < 8192; i += 256) {
        int b = i >> 10, j = i & 1023; float a = s2b[j];
        for (int k = 0; k < 64; k++) a += t3[b * 64 + k] * s2w[k * 1024 + j];
        if (j < 512) { float sp = (a > 20.f) ? a : log1pf(expf(a)); gamma[b * 512 + j] = sp + 1e-8f; }
        else beta[b * 512 + (j - 512)] = a;
    }
}

// ---------------- modulation: hf_bf16 = (h - mu)*rsig*gamma + beta ----------------
__global__ __launch_bounds__(256) void modulate_kernel(const float* __restrict__ h,
                                                       const float* __restrict__ mu,
                                                       const float* __restrict__ rsig,
                                                       const float* __restrict__ gamma,
                                                       const float* __restrict__ beta,
                                                       u16* __restrict__ hf) {
    int i = blockIdx.x * 256 + threadIdx.x;     // 4194304 elements
    int d = i & 511; int n = i >> 9; int b = n >> 10;
    int bd = b * 512 + d;
    float v = (h[i] - mu[bd]) * rsig[bd] * gamma[bd] + beta[bd];
    hf[i] = f2bf(v);
}

// ---------------- gate GEMM: g1h = relu(hf @ g1 + b1)  (M=8192,N=256,K=512) ----------------
__global__ __launch_bounds__(256) void gemm_gate_kernel(const u16* __restrict__ A,   // [8192][512]
                                                        const u16* __restrict__ BT,  // [256][512]
                                                        const float* __restrict__ bias,
                                                        u16* __restrict__ Cout) {    // [8192][256]
    __shared__ __align__(16) u16 As[2][128 * LDT];
    __shared__ __align__(16) u16 Bt[2][64 * LDT];
    const int tid = threadIdx.x;
    const int mBase = blockIdx.y * 128, nBase = blockIdx.x * 64;
    const int ar = tid >> 1, ap = (tid & 1) * 16;
    const int br = tid >> 2, bp = (tid & 3) * 8;
    const u16* aG = A + (size_t)(mBase + ar) * 512 + ap;
    const u16* bG = BT + (size_t)(nBase + br) * 512 + bp;
    const int lane = tid & 31, lr = lane & 15, lhi = lane >> 4;
    const int wid = tid >> 5, wy = wid & 3, wx = wid >> 2;

    f32x8 acc[2][2];
    f32x8 z = {0.f,0.f,0.f,0.f,0.f,0.f,0.f,0.f};
#pragma unroll
    for (int mf = 0; mf < 2; mf++)
#pragma unroll
        for (int nf = 0; nf < 2; nf++) acc[mf][nf] = z;

#if USE_ASYNC
    async_cp16(aG, &As[0][ar * LDT + ap]);
    async_cp16(aG + 8, &As[0][ar * LDT + ap + 8]);
    async_cp16(bG, &Bt[0][br * LDT + bp]);
#endif
    for (int kk = 0; kk < 512; kk += 32) {
        const int cur = (kk >> 5) & 1;
#if USE_ASYNC
        wait_async0();
        __syncthreads();
        if (kk + 32 < 512) {
            const int nxt = 1 - cur;
            async_cp16(aG + kk + 32, &As[nxt][ar * LDT + ap]);
            async_cp16(aG + kk + 40, &As[nxt][ar * LDT + ap + 8]);
            async_cp16(bG + kk + 32, &Bt[nxt][br * LDT + bp]);
        }
#else
        const uint4* pa = reinterpret_cast<const uint4*>(aG + kk);
        uint4 a0 = pa[0], a1 = pa[1];
        uint4 b0 = *reinterpret_cast<const uint4*>(bG + kk);
        __syncthreads();
        *reinterpret_cast<uint4*>(&As[cur][ar * LDT + ap]) = a0;
        *reinterpret_cast<uint4*>(&As[cur][ar * LDT + ap + 8]) = a1;
        *reinterpret_cast<uint4*>(&Bt[cur][br * LDT + bp]) = b0;
        __syncthreads();
#endif
        bf16x16 af[2], bf[2];
#pragma unroll
        for (int mf = 0; mf < 2; mf++) af[mf] = frag_a(As[cur], (wy * 32 + mf * 16 + lr) * LDT + lhi * 16);
#pragma unroll
        for (int nf = 0; nf < 2; nf++) bf[nf] = frag_b(Bt[cur], (wx * 32 + nf * 16 + lr) * LDT + lhi * 16);
#pragma unroll
        for (int mf = 0; mf < 2; mf++)
#pragma unroll
            for (int nf = 0; nf < 2; nf++) acc[mf][nf] = wmma_bf16(af[mf], bf[nf], acc[mf][nf]);
    }
#pragma unroll
    for (int mf = 0; mf < 2; mf++)
#pragma unroll
        for (int nf = 0; nf < 2; nf++) {
            int colb = nBase + wx * 32 + nf * 16 + lr;
            int rowb = mBase + wy * 32 + mf * 16 + lhi * 8;
            float bv = bias[colb];
#pragma unroll
            for (int v = 0; v < 8; v++) {
                float val = acc[mf][nf][v] + bv;
                Cout[(size_t)(rowb + v) * 256 + colb] = f2bf(val > 0.f ? val : 0.f);
            }
        }
}

// ---------------- gate part 2: logits, softmax, top-2, counts ----------------
__global__ __launch_bounds__(256) void gate2_kernel(const u16* __restrict__ g1h,
                                                    const float* __restrict__ g2w,  // [256][8]
                                                    const float* __restrict__ g2b,
                                                    int* __restrict__ eids, float* __restrict__ gvals,
                                                    int* __restrict__ counts) {
    __shared__ float g2s[2048];
    for (int i = threadIdx.x; i < 2048; i += 256) g2s[i] = g2w[i];
    __syncthreads();
    int n = blockIdx.x * 256 + threadIdx.x;   // 8192 tokens
    float acc[8];
#pragma unroll
    for (int e = 0; e < 8; e++) acc[e] = g2b[e];
    const u16* row = g1h + (size_t)n * 256;
    for (int k = 0; k < 256; k++) {
        float hv = bf2f(row[k]);
#pragma unroll
        for (int e = 0; e < 8; e++) acc[e] += hv * g2s[k * 8 + e];
    }
    float m = acc[0];
#pragma unroll
    for (int e = 1; e < 8; e++) m = fmaxf(m, acc[e]);
    float p[8]; float sum = 0.f;
#pragma unroll
    for (int e = 0; e < 8; e++) { p[e] = expf(acc[e] - m); sum += p[e]; }
    float inv = 1.f / sum;
#pragma unroll
    for (int e = 0; e < 8; e++) p[e] *= inv;
    int e0 = 0; float p0 = p[0];
#pragma unroll
    for (int e = 1; e < 8; e++) if (p[e] > p0) { p0 = p[e]; e0 = e; }
    int e1 = (e0 == 0) ? 1 : 0; float p1 = p[e1];
#pragma unroll
    for (int e = 0; e < 8; e++) if (e != e0 && p[e] > p1) { p1 = p[e]; e1 = e; }
    float ws = 1.f / (p0 + p1 + 1e-8f);
    eids[n * 2] = e0; eids[n * 2 + 1] = e1;
    gvals[n * 2] = p0 * ws; gvals[n * 2 + 1] = p1 * ws;
    atomicAdd(&counts[e0], 1); atomicAdd(&counts[e1], 1);
}

// ---------------- routing bookkeeping ----------------
__global__ __launch_bounds__(256) void init_route_kernel(int* counts, int* cursor, int* rowidx,
                                                         u16* zrow) {
    int i = blockIdx.x * 256 + threadIdx.x;
    if (i < 17408) rowidx[i] = -1;
    if (i < 512) zrow[i] = 0;
    if (i < 8) { counts[i] = 0; cursor[i] = 0; }
}

__global__ void scan_kernel(const int* __restrict__ counts, int* __restrict__ offsets,
                            int* __restrict__ tile_expert) {
    if (threadIdx.x == 0) {
        int off = 0;
        for (int e = 0; e < 8; e++) {
            offsets[e] = off;
            int c = counts[e];
            int cp = (c + 127) & ~127;
            for (int tm = off >> 7; tm < (off + cp) >> 7; ++tm) tile_expert[tm] = e;
            off += cp;
        }
        offsets[8] = off;
        for (int tm = off >> 7; tm < 136; ++tm) tile_expert[tm] = 0;
    }
}

__global__ __launch_bounds__(256) void scatter_kernel(const int* __restrict__ eids,
                                                      const int* __restrict__ offsets,
                                                      int* __restrict__ cursor,
                                                      int* __restrict__ rowidx,
                                                      int* __restrict__ posmap) {
    int n = blockIdx.x * 256 + threadIdx.x;   // 8192
#pragma unroll
    for (int k = 0; k < 2; k++) {
        int e = eids[n * 2 + k];
        int pos = offsets[e] + atomicAdd(&cursor[e], 1);
        rowidx[pos] = n;
        posmap[n * 2 + k] = pos;
    }
}

// ---------------- expert GEMM 1: hid = relu(A@wi0) * (A@wi1), A gathered by rowidx ----------------
__global__ __launch_bounds__(256) void gemm_dual_kernel(const u16* __restrict__ hf,   // [8192][512]
                                                        const u16* __restrict__ w0T,  // [E][1024][512]
                                                        const u16* __restrict__ w1T,
                                                        const int* __restrict__ rowidx,
                                                        const int* __restrict__ tile_expert,
                                                        const u16* __restrict__ zrow, // 512 zero halfs
                                                        u16* __restrict__ hid) {      // [P][1024]
    __shared__ __align__(16) u16 As[2][128 * LDT];
    __shared__ __align__(16) u16 B0[2][64 * LDT];
    __shared__ __align__(16) u16 B1[2][64 * LDT];
    const int tid = threadIdx.x;
    const int mBase = blockIdx.y * 128, nBase = blockIdx.x * 64;
    const int e = tile_expert[blockIdx.y];
    const u16* b0g = w0T + (size_t)e * 1024 * 512;
    const u16* b1g = w1T + (size_t)e * 1024 * 512;
    const int ar = tid >> 1, ap = (tid & 1) * 16;
    const int br = tid >> 2, bp = (tid & 3) * 8;
    const int tok = rowidx[mBase + ar];
    const u16* aG = (tok >= 0) ? (hf + (size_t)tok * 512 + ap) : (zrow + ap);
    const u16* b0p = b0g + (size_t)(nBase + br) * 512 + bp;
    const u16* b1p = b1g + (size_t)(nBase + br) * 512 + bp;
    const int aStep = (tok >= 0) ? 1 : 0;   // pad rows keep reading the zero row
    const int lane = tid & 31, lr = lane & 15, lhi = lane >> 4;
    const int wid = tid >> 5, wy = wid & 3, wx = wid >> 2;

    f32x8 c0[2][2], c1[2][2];
    f32x8 z = {0.f,0.f,0.f,0.f,0.f,0.f,0.f,0.f};
#pragma unroll
    for (int mf = 0; mf < 2; mf++)
#pragma unroll
        for (int nf = 0; nf < 2; nf++) { c0[mf][nf] = z; c1[mf][nf] = z; }

#if USE_ASYNC
    async_cp16(aG, &As[0][ar * LDT + ap]);
    async_cp16(aG + 8 * aStep, &As[0][ar * LDT + ap + 8]);
    async_cp16(b0p, &B0[0][br * LDT + bp]);
    async_cp16(b1p, &B1[0][br * LDT + bp]);
#endif
    for (int kk = 0; kk < 512; kk += 32) {
        const int cur = (kk >> 5) & 1;
#if USE_ASYNC
        wait_async0();
        __syncthreads();
        if (kk + 32 < 512) {
            const int nxt = 1 - cur;
            async_cp16(aG + (kk + 32) * aStep, &As[nxt][ar * LDT + ap]);
            async_cp16(aG + (kk + 40) * aStep, &As[nxt][ar * LDT + ap + 8]);
            async_cp16(b0p + kk + 32, &B0[nxt][br * LDT + bp]);
            async_cp16(b1p + kk + 32, &B1[nxt][br * LDT + bp]);
        }
#else
        const uint4* pa = reinterpret_cast<const uint4*>(aG + kk * aStep);
        uint4 a0 = pa[0];
        uint4 a1 = *reinterpret_cast<const uint4*>(aG + (kk + 8) * aStep);
        uint4 v0 = *reinterpret_cast<const uint4*>(b0p + kk);
        uint4 v1 = *reinterpret_cast<const uint4*>(b1p + kk);
        __syncthreads();
        *reinterpret_cast<uint4*>(&As[cur][ar * LDT + ap]) = a0;
        *reinterpret_cast<uint4*>(&As[cur][ar * LDT + ap + 8]) = a1;
        *reinterpret_cast<uint4*>(&B0[cur][br * LDT + bp]) = v0;
        *reinterpret_cast<uint4*>(&B1[cur][br * LDT + bp]) = v1;
        __syncthreads();
#endif
        bf16x16 af[2], b0f[2], b1f[2];
#pragma unroll
        for (int mf = 0; mf < 2; mf++) af[mf] = frag_a(As[cur], (wy * 32 + mf * 16 + lr) * LDT + lhi * 16);
#pragma unroll
        for (int nf = 0; nf < 2; nf++) {
            b0f[nf] = frag_b(B0[cur], (wx * 32 + nf * 16 + lr) * LDT + lhi * 16);
            b1f[nf] = frag_b(B1[cur], (wx * 32 + nf * 16 + lr) * LDT + lhi * 16);
        }
#pragma unroll
        for (int mf = 0; mf < 2; mf++)
#pragma unroll
            for (int nf = 0; nf < 2; nf++) {
                c0[mf][nf] = wmma_bf16(af[mf], b0f[nf], c0[mf][nf]);
                c1[mf][nf] = wmma_bf16(af[mf], b1f[nf], c1[mf][nf]);
            }
    }
#pragma unroll
    for (int mf = 0; mf < 2; mf++)
#pragma unroll
        for (int nf = 0; nf < 2; nf++) {
            int colb = nBase + wx * 32 + nf * 16 + lr;
            int rowb = mBase + wy * 32 + mf * 16 + lhi * 8;
#pragma unroll
            for (int v = 0; v < 8; v++) {
                float r0 = c0[mf][nf][v]; r0 = r0 > 0.f ? r0 : 0.f;
                hid[(size_t)(rowb + v) * 1024 + colb] = f2bf(r0 * c1[mf][nf][v]);
            }
        }
}

// ---------------- expert GEMM 2: eo = hid @ wo[e]   (K=1024, N=512) ----------------
__global__ __launch_bounds__(256) void gemm_eo_kernel(const u16* __restrict__ hid,  // [P][1024]
                                                      const u16* __restrict__ woT,  // [E][512][1024]
                                                      const int* __restrict__ tile_expert,
                                                      float* __restrict__ eo) {     // [P][512]
    __shared__ __align__(16) u16 As[2][128 * LDT];
    __shared__ __align__(16) u16 Bt[2][64 * LDT];
    const int tid = threadIdx.x;
    const int mBase = blockIdx.y * 128, nBase = blockIdx.x * 64;
    const int e = tile_expert[blockIdx.y];
    const int ar = tid >> 1, ap = (tid & 1) * 16;
    const int br = tid >> 2, bp = (tid & 3) * 8;
    const u16* aG = hid + (size_t)(mBase + ar) * 1024 + ap;
    const u16* bG = woT + (size_t)e * 512 * 1024 + (size_t)(nBase + br) * 1024 + bp;
    const int lane = tid & 31, lr = lane & 15, lhi = lane >> 4;
    const int wid = tid >> 5, wy = wid & 3, wx = wid >> 2;

    f32x8 acc[2][2];
    f32x8 z = {0.f,0.f,0.f,0.f,0.f,0.f,0.f,0.f};
#pragma unroll
    for (int mf = 0; mf < 2; mf++)
#pragma unroll
        for (int nf = 0; nf < 2; nf++) acc[mf][nf] = z;

#if USE_ASYNC
    async_cp16(aG, &As[0][ar * LDT + ap]);
    async_cp16(aG + 8, &As[0][ar * LDT + ap + 8]);
    async_cp16(bG, &Bt[0][br * LDT + bp]);
#endif
    for (int kk = 0; kk < 1024; kk += 32) {
        const int cur = (kk >> 5) & 1;
#if USE_ASYNC
        wait_async0();
        __syncthreads();
        if (kk + 32 < 1024) {
            const int nxt = 1 - cur;
            async_cp16(aG + kk + 32, &As[nxt][ar * LDT + ap]);
            async_cp16(aG + kk + 40, &As[nxt][ar * LDT + ap + 8]);
            async_cp16(bG + kk + 32, &Bt[nxt][br * LDT + bp]);
        }
#else
        const uint4* pa = reinterpret_cast<const uint4*>(aG + kk);
        uint4 a0 = pa[0], a1 = pa[1];
        uint4 b0 = *reinterpret_cast<const uint4*>(bG + kk);
        __syncthreads();
        *reinterpret_cast<uint4*>(&As[cur][ar * LDT + ap]) = a0;
        *reinterpret_cast<uint4*>(&As[cur][ar * LDT + ap + 8]) = a1;
        *reinterpret_cast<uint4*>(&Bt[cur][br * LDT + bp]) = b0;
        __syncthreads();
#endif
        bf16x16 af[2], bf[2];
#pragma unroll
        for (int mf = 0; mf < 2; mf++) af[mf] = frag_a(As[cur], (wy * 32 + mf * 16 + lr) * LDT + lhi * 16);
#pragma unroll
        for (int nf = 0; nf < 2; nf++) bf[nf] = frag_b(Bt[cur], (wx * 32 + nf * 16 + lr) * LDT + lhi * 16);
#pragma unroll
        for (int mf = 0; mf < 2; mf++)
#pragma unroll
            for (int nf = 0; nf < 2; nf++) acc[mf][nf] = wmma_bf16(af[mf], bf[nf], acc[mf][nf]);
    }
#pragma unroll
    for (int mf = 0; mf < 2; mf++)
#pragma unroll
        for (int nf = 0; nf < 2; nf++) {
            int colb = nBase + wx * 32 + nf * 16 + lr;
            int rowb = mBase + wy * 32 + mf * 16 + lhi * 8;
#pragma unroll
            for (int v = 0; v < 8; v++)
                eo[(size_t)(rowb + v) * 512 + colb] = acc[mf][nf][v];
        }
}

// ---------------- combine: y = x + w0*eo[p0] + w1*eo[p1]  (deterministic gather) ----------------
__global__ __launch_bounds__(256) void combine_kernel(const float* __restrict__ xin,
                                                      const float* __restrict__ eo,
                                                      const int* __restrict__ posmap,
                                                      const float* __restrict__ gvals,
                                                      float* __restrict__ y) {
    int i = blockIdx.x * 256 + threadIdx.x;   // 4194304
    int d = i & 511; int n = i >> 9;
    int p0 = posmap[n * 2], p1 = posmap[n * 2 + 1];
    float w0 = gvals[n * 2], w1 = gvals[n * 2 + 1];
    y[i] = xin[i] + w0 * eo[(size_t)p0 * 512 + d] + w1 * eo[(size_t)p1 * 512 + d];
}

// ---------------- host ----------------
extern "C" void kernel_launch(void* const* d_in, const int* in_sizes, int n_in,
                              void* d_out, int out_size, void* d_ws, size_t ws_size,
                              hipStream_t stream) {
    const float* x0   = (const float*)d_in[0];
    const int*   sid  = (const int*)d_in[1];
    const float* ln_g = (const float*)d_in[2];
    const float* ln_b = (const float*)d_in[3];
    const float* semb = (const float*)d_in[4];
    const float* h1w  = (const float*)d_in[5];
    const float* h1b  = (const float*)d_in[6];
    const float* h2w  = (const float*)d_in[7];
    const float* h2b  = (const float*)d_in[8];
    const float* s1w  = (const float*)d_in[9];
    const float* s1b  = (const float*)d_in[10];
    const float* s2w  = (const float*)d_in[11];
    const float* s2b  = (const float*)d_in[12];
    const float* g1w  = (const float*)d_in[13];
    const float* g1b  = (const float*)d_in[14];
    const float* g2w  = (const float*)d_in[15];
    const float* g2b  = (const float*)d_in[16];
    const float* wi0  = (const float*)d_in[17];
    const float* wi1  = (const float*)d_in[18];
    const float* wo   = (const float*)d_in[19];
    (void)in_sizes; (void)n_in; (void)out_size; (void)ws_size;

    char* ws = (char*)d_ws;
    size_t off = 0;
    auto alloc = [&](size_t bytes) -> char* {
        char* p = ws + off; off += (bytes + 255) & ~(size_t)255; return p;
    };
    float* h     = (float*)alloc((size_t)4194304 * 4);
    float* bufA  = (float*)alloc((size_t)4194304 * 4);
    float* mu    = (float*)alloc(4096 * 4);
    float* rsig  = (float*)alloc(4096 * 4);
    float* gamma = (float*)alloc(4096 * 4);
    float* beta  = (float*)alloc(4096 * 4);
    u16*  hfb    = (u16*)alloc((size_t)4194304 * 2);
    u16*  g1h    = (u16*)alloc((size_t)8192 * 256 * 2);
    int*  eids   = (int*)alloc(16384 * 4);
    float* gv    = (float*)alloc(16384 * 4);
    int*  posmap = (int*)alloc(16384 * 4);
    int*  counts = (int*)alloc(64);
    int*  curs   = (int*)alloc(64);
    int*  offs   = (int*)alloc(64);
    int*  texp   = (int*)alloc(1024);
    int*  rowidx = (int*)alloc(17408 * 4);
    u16*  zrow   = (u16*)alloc(512 * 2);
    u16*  hid    = (u16*)alloc((size_t)17408 * 1024 * 2);
    float* eo    = (float*)alloc((size_t)17408 * 512 * 4);
    u16*  w0T    = (u16*)alloc((size_t)4194304 * 2);
    u16*  w1T    = (u16*)alloc((size_t)4194304 * 2);
    u16*  woT    = (u16*)alloc((size_t)4194304 * 2);
    u16*  g1T    = (u16*)alloc((size_t)131072 * 2);

    // one-time (per launch) weight conversion + transpose to bf16 N-major
    convT_kernel<<<16384, 256, 0, stream>>>(wi0, w0T, 512, 1024, 4194304L);
    convT_kernel<<<16384, 256, 0, stream>>>(wi1, w1T, 512, 1024, 4194304L);
    convT_kernel<<<16384, 256, 0, stream>>>(wo,  woT, 1024, 512, 4194304L);
    convT_kernel<<<512,   256, 0, stream>>>(g1w, g1T, 512, 256, 131072L);

    for (int l = 0; l < 2; l++) {
        const float* xin = (l == 0) ? x0 : bufA;
        float* yout = (l == 0) ? bufA : (float*)d_out;
        ln_kernel<<<8192, 128, 0, stream>>>(xin, ln_g + l * 512, ln_b + l * 512, h);
        colstats_kernel<<<dim3(8, 8), 256, 0, stream>>>(h, mu, rsig);
        subj_kernel<<<1, 256, 0, stream>>>(sid, semb + (size_t)l * 64 * 64,
                                           h1w + l * 64 * 128, h1b + l * 128,
                                           h2w + l * 128 * 128, h2b + l * 128,
                                           s1w + l * 128 * 64, s1b + l * 64,
                                           s2w + l * 64 * 1024, s2b + l * 1024,
                                           gamma, beta);
        modulate_kernel<<<16384, 256, 0, stream>>>(h, mu, rsig, gamma, beta, hfb);
        init_route_kernel<<<68, 256, 0, stream>>>(counts, curs, rowidx, zrow);
        gemm_gate_kernel<<<dim3(4, 64), 256, 0, stream>>>(hfb, g1T, g1b, g1h);
        gate2_kernel<<<32, 256, 0, stream>>>(g1h, g2w, g2b, eids, gv, counts);
        scan_kernel<<<1, 32, 0, stream>>>(counts, offs, texp);
        scatter_kernel<<<32, 256, 0, stream>>>(eids, offs, curs, rowidx, posmap);
        gemm_dual_kernel<<<dim3(16, 136), 256, 0, stream>>>(hfb, w0T, w1T, rowidx, texp, zrow, hid);
        gemm_eo_kernel<<<dim3(8, 136), 256, 0, stream>>>(hid, woT, texp, eo);
        combine_kernel<<<16384, 256, 0, stream>>>(xin, eo, posmap, gv, yout);
    }
}
